// ResMamba_27384711479946
// MI455X (gfx1250) — matrix operand order repaired and checked by reading, not compiled, not run on hardware
//
#include <hip/hip_runtime.h>

#define BB 2
#define LL 2048
#define DMODEL 256
#define DSTATE 128
#define DCONV 4
#define HEADDIM 64
#define DINNER 1024
#define NHEADS 16
#define CONVDIM 1280
#define DINPROJ 2320
#define NCHUNK 8
#define CHUNKL 256
#define MTOT (BB*LL)
#define GN_EPS 1.1920928955078125e-07f
#define RMS_EPS 1e-5f

typedef __bf16 bf16_t;
typedef bf16_t bf16x16 __attribute__((ext_vector_type(16)));
typedef float  f32x8   __attribute__((ext_vector_type(8)));
typedef unsigned short u16;

__device__ __forceinline__ int lane_id() { return (int)(threadIdx.x & 31u); }
__device__ __forceinline__ int wave_id() { return (int)(threadIdx.x >> 5u); }

__device__ __forceinline__ u16 f2bf(float f) {
  unsigned u = __float_as_uint(f);
  u += 0x7FFFu + ((u >> 16) & 1u);           // round-to-nearest-even
  return (u16)(u >> 16);
}
__device__ __forceinline__ float siluf(float x) { return x / (1.f + expf(-x)); }
__device__ __forceinline__ float softplusf(float x) { return (x > 20.f) ? x : log1pf(expf(x)); }

// A fragment: 16x32 bf16, lane = M, K contiguous k = 16*(lane/16) + j (LDS layout [m][k], ld = k-stride)
__device__ __forceinline__ bf16x16 frag_a(const u16* s, int m0, int ld, int k0) {
  int l = lane_id();
  int m = m0 + (l & 15);
  int kh = (l < 16) ? 0 : 16;
  bf16x16 a;
#pragma unroll
  for (int j = 0; j < 16; ++j) {
    int k = k0 + kh + j;
    a[j] = __builtin_bit_cast(bf16_t, s[m * ld + k]);
  }
  return a;
}
// B fragment from [k][n] LDS tile (ld = n-stride)
__device__ __forceinline__ bf16x16 frag_b_k(const u16* s, int k0, int ld, int n0) {
  int l = lane_id();
  int n = n0 + (l & 15);
  int kh = (l < 16) ? 0 : 16;
  bf16x16 bv;
#pragma unroll
  for (int j = 0; j < 16; ++j)
    bv[j] = __builtin_bit_cast(bf16_t, s[(k0 + kh + j) * ld + n]);
  return bv;
}
// B fragment from [n][k] LDS tile (ld = k-stride) -> 16 contiguous bf16 per lane (2x b128)
__device__ __forceinline__ bf16x16 frag_b_n(const u16* s, int n0, int ld, int k0) {
  int l = lane_id();
  int n = n0 + (l & 15);
  int kh = (l < 16) ? 0 : 16;
  bf16x16 bv;
#pragma unroll
  for (int j = 0; j < 16; ++j)
    bv[j] = __builtin_bit_cast(bf16_t, s[n * ld + k0 + kh + j]);
  return bv;
}
__device__ __forceinline__ f32x8 wmma_bf16(bf16x16 a, bf16x16 b, f32x8 c) {
  return __builtin_amdgcn_wmma_f32_16x16x32_bf16(false, a, false, b, (short)0, c, false, false);
}

// ---------------- elementwise / reduction kernels ----------------

__global__ void k_tobf16(const float* __restrict__ s, u16* __restrict__ d, int n) {
  int i = blockIdx.x * 256 + threadIdx.x;
  if (i < n) d[i] = f2bf(s[i]);
}

__global__ void k_gn_stats(const float* __restrict__ x, float* __restrict__ stats) {
  int b = blockIdx.x;
  const float* p = x + (size_t)b * DMODEL * LL;
  double s = 0.0, s2 = 0.0;
  for (int i = threadIdx.x; i < DMODEL * LL; i += 256) {
    float v = p[i];
    s += (double)v; s2 += (double)v * (double)v;
  }
  __shared__ double rs[256], rs2[256];
  rs[threadIdx.x] = s; rs2[threadIdx.x] = s2;
  __syncthreads();
  for (int o = 128; o > 0; o >>= 1) {
    if ((int)threadIdx.x < o) { rs[threadIdx.x] += rs[threadIdx.x + o]; rs2[threadIdx.x] += rs2[threadIdx.x + o]; }
    __syncthreads();
  }
  if (threadIdx.x == 0) {
    double n = (double)(DMODEL * LL);
    double mu = rs[0] / n;
    double var = rs2[0] / n - mu * mu;
    stats[b * 2 + 0] = (float)mu;
    stats[b * 2 + 1] = rsqrtf((float)var + GN_EPS);
  }
}

__global__ void k_normT(const float* __restrict__ x, const float* __restrict__ stats,
                        const float* __restrict__ gw, const float* __restrict__ gb,
                        float* __restrict__ t, u16* __restrict__ tb) {
  int idx = blockIdx.x * 256 + threadIdx.x;        // (b*LL + l)*DMODEL + d
  int b = idx >> 19;
  int rem = idx & ((1 << 19) - 1);
  int l = rem >> 8, d = rem & 255;
  float v = x[((size_t)b * DMODEL + d) * LL + l];
  float o = (v - stats[2 * b]) * stats[2 * b + 1] * gw[d] + gb[d];
  t[idx] = o;
  tb[idx] = f2bf(o);
}

__global__ void k_conv(const float* __restrict__ zx, const float* __restrict__ cw,
                       const float* __restrict__ cbias, float* __restrict__ xbc,
                       u16* __restrict__ bcb) {
  int idx = blockIdx.x * 256 + threadIdx.x;
  if (idx >= MTOT * CONVDIM) return;
  int row = idx / CONVDIM, ch = idx % CONVDIM;
  int b = row >> 11, l = row & (LL - 1);
  float s = cbias[ch];
#pragma unroll
  for (int j = 0; j < DCONV; ++j) {
    int li = l - (DCONV - 1) + j;
    if (li >= 0) s += cw[ch * DCONV + j] * zx[(size_t)(b * LL + li) * DINPROJ + DINNER + ch];
  }
  float v = siluf(s);
  xbc[idx] = v;
  if (ch >= DINNER) bcb[(size_t)row * 256 + (ch - DINNER)] = f2bf(v);  // [B | C] bf16
}

__global__ void k_dt_scan(const float* __restrict__ zx, const float* __restrict__ dtbias,
                          const float* __restrict__ alog, float* __restrict__ dtb,
                          float* __restrict__ cA) {
  __shared__ float sc[CHUNKL];
  int blk = blockIdx.x;                            // b*(H*NC) + h*NC + c
  int b = blk / (NHEADS * NCHUNK);
  int rem = blk % (NHEADS * NCHUNK);
  int h = rem / NCHUNK, c = rem % NCHUNK;
  int l = c * CHUNKL + threadIdx.x;
  float raw = zx[(size_t)(b * LL + l) * DINPROJ + (DINPROJ - NHEADS) + h];
  float dt = softplusf(raw + dtbias[h]);
  dtb[(size_t)(b * LL + l) * NHEADS + h] = dt;
  float a = -expf(alog[h]) * dt;
  sc[threadIdx.x] = a;
  __syncthreads();
  for (int o = 1; o < CHUNKL; o <<= 1) {           // inclusive scan
    float v = ((int)threadIdx.x >= o) ? sc[threadIdx.x - o] : 0.f;
    __syncthreads();
    sc[threadIdx.x] += v;
    __syncthreads();
  }
  cA[(size_t)(b * NHEADS + h) * LL + l] = sc[threadIdx.x];
}

__global__ void k_xs(const float* __restrict__ xbc, const float* __restrict__ dtb,
                     float* __restrict__ xs, u16* __restrict__ xsb) {
  int idx = blockIdx.x * 256 + threadIdx.x;
  int row = idx >> 10, i = idx & 1023;
  float v = dtb[(size_t)row * NHEADS + (i >> 6)] * xbc[(size_t)row * CONVDIM + i];
  xs[idx] = v;
  xsb[idx] = f2bf(v);
}

__global__ void k_chunkrec(const float* __restrict__ st, const float* __restrict__ cA,
                           u16* __restrict__ nsb) {
  int g = blockIdx.x * 256 + threadIdx.x;
  if (g >= BB * NHEADS * HEADDIM * DSTATE) return;
  int b = g / (NHEADS * HEADDIM * DSTATE);
  int rem = g - b * (NHEADS * HEADDIM * DSTATE);
  int h = rem / (HEADDIM * DSTATE);
  int pn = rem % (HEADDIM * DSTATE);
  const float* ca = cA + (size_t)(b * NHEADS + h) * LL;
  float run = 0.f;
  nsb[(size_t)((b * NCHUNK + 0) * NHEADS + h) * HEADDIM * DSTATE + pn] = 0;
  for (int z = 1; z < NCHUNK; ++z) {
    float sp = st[(size_t)((b * NCHUNK + z - 1) * NHEADS + h) * HEADDIM * DSTATE + pn];
    run = sp + expf(ca[(z - 1) * CHUNKL + CHUNKL - 1]) * run;
    nsb[(size_t)((b * NCHUNK + z) * NHEADS + h) * HEADDIM * DSTATE + pn] = f2bf(run);
  }
}

__global__ void k_gate_rms(const float* __restrict__ Y, const float* __restrict__ xbc,
                           const float* __restrict__ zx, const float* __restrict__ Dh,
                           const float* __restrict__ nw, u16* __restrict__ ynb) {
  __shared__ float red[256];
  int row = blockIdx.x;
  float v[4]; float ss = 0.f;
#pragma unroll
  for (int i = 0; i < 4; ++i) {
    int d = threadIdx.x + i * 256;
    float yv = Y[(size_t)row * DINNER + d] + xbc[(size_t)row * CONVDIM + d] * Dh[d >> 6];
    float zv = zx[(size_t)row * DINPROJ + d];
    float g = yv * siluf(zv);
    v[i] = g; ss += g * g;
  }
  red[threadIdx.x] = ss;
  __syncthreads();
  for (int o = 128; o > 0; o >>= 1) {
    if ((int)threadIdx.x < o) red[threadIdx.x] += red[threadIdx.x + o];
    __syncthreads();
  }
  float scale = rsqrtf(red[0] * (1.f / DINNER) + RMS_EPS);
#pragma unroll
  for (int i = 0; i < 4; ++i) {
    int d = threadIdx.x + i * 256;
    ynb[(size_t)row * DINNER + d] = f2bf(v[i] * scale * nw[d]);
  }
}

// ---------------- WMMA GEMM kernels ----------------
// 64x128 tile: 8 waves 2(M)x4(N), each wave 32x32 (2x2 WMMA accs)
// 64x64  tile: 8 waves 4(M)x2(N), each wave 16x32 (1x2 WMMA accs)

// double-buffered: C[m,n] = sum_k tb[m,k] * wb[n,k] ; M=4096, N=2320, K=256
__global__ void k_in_proj(const u16* __restrict__ tb, const u16* __restrict__ wb,
                          float* __restrict__ out, int flip) {
  __shared__ __align__(16) u16 sA[2][64 * 32];
  __shared__ __align__(16) u16 sB[2][128 * 32];
  const int m0 = blockIdx.x * 64, n0 = blockIdx.y * 128;
  const int wv = wave_id(), wm = wv >> 2, wn = wv & 3;
  const int tid = threadIdx.x;

  const int ar = tid >> 2, aseg = tid & 3;
  int gmA = m0 + ar, bA = gmA >> 11, lA = gmA & (LL - 1);
  int lsA = flip ? (LL - 1 - lA) : lA;
  const u16* arow = tb + (size_t)(bA * LL + lsA) * DMODEL + aseg * 8;
  const int bn = tid >> 2, bseg = tid & 3;
  int gn0 = n0 + bn, gn1 = n0 + 64 + bn;
  const u16* brow0 = wb + (size_t)gn0 * DMODEL + bseg * 8;
  const u16* brow1 = wb + (size_t)gn1 * DMODEL + bseg * 8;
  const uint4 zz = make_uint4(0, 0, 0, 0);

  f32x8 acc[2][2] = {};
  uint4 ra  = *(const uint4*)(arow);
  uint4 rb0 = (gn0 < DINPROJ) ? *(const uint4*)(brow0) : zz;
  uint4 rb1 = (gn1 < DINPROJ) ? *(const uint4*)(brow1) : zz;
  *(uint4*)&sA[0][ar * 32 + aseg * 8] = ra;
  *(uint4*)&sB[0][bn * 32 + bseg * 8] = rb0;
  *(uint4*)&sB[0][(64 + bn) * 32 + bseg * 8] = rb1;
  __syncthreads();

  const int STEPS = DMODEL / 32;
  for (int s = 0; s < STEPS; ++s) {
    int cur = s & 1;
    if (s + 1 < STEPS) {
      int k0 = (s + 1) * 32;
      ra  = *(const uint4*)(arow + k0);
      rb0 = (gn0 < DINPROJ) ? *(const uint4*)(brow0 + k0) : zz;
      rb1 = (gn1 < DINPROJ) ? *(const uint4*)(brow1 + k0) : zz;
    }
#pragma unroll
    for (int i = 0; i < 2; ++i) {
      bf16x16 a = frag_a(sA[cur], wm * 32 + i * 16, 32, 0);
#pragma unroll
      for (int j = 0; j < 2; ++j) {
        bf16x16 bb = frag_b_n(sB[cur], wn * 32 + j * 16, 32, 0);
        acc[i][j] = wmma_bf16(a, bb, acc[i][j]);
      }
    }
    if (s + 1 < STEPS) {
      int nx = 1 - cur;
      *(uint4*)&sA[nx][ar * 32 + aseg * 8] = ra;
      *(uint4*)&sB[nx][bn * 32 + bseg * 8] = rb0;
      *(uint4*)&sB[nx][(64 + bn) * 32 + bseg * 8] = rb1;
    }
    __syncthreads();
  }
  int l = lane_id(); int mlo = (l < 16) ? 0 : 8, nb = l & 15;
#pragma unroll
  for (int i = 0; i < 2; ++i)
#pragma unroll
    for (int j = 0; j < 2; ++j)
#pragma unroll
      for (int r = 0; r < 8; ++r) {
        int gm = m0 + wm * 32 + i * 16 + mlo + r;
        int gn = n0 + wn * 32 + j * 16 + nb;
        if (gn < DINPROJ) out[(size_t)gm * DINPROJ + gn] = acc[i][j][r];
      }
}

// double-buffered: out[m,n] = sum_k ynb[m,k] * owb[n,k] ; M=4096, N=256, K=1024
__global__ void k_outproj(const u16* __restrict__ yb, const u16* __restrict__ ob,
                          float* __restrict__ out) {
  __shared__ __align__(16) u16 sA[2][64 * 32];
  __shared__ __align__(16) u16 sB[2][128 * 32];
  const int m0 = blockIdx.x * 64, n0 = blockIdx.y * 128;
  const int wv = wave_id(), wm = wv >> 2, wn = wv & 3;
  const int tid = threadIdx.x;
  const int ar = tid >> 2, aseg = tid & 3;
  const u16* arow = yb + (size_t)(m0 + ar) * DINNER + aseg * 8;
  const int bn = tid >> 2, bseg = tid & 3;
  const u16* brow0 = ob + (size_t)(n0 + bn) * DINNER + bseg * 8;
  const u16* brow1 = ob + (size_t)(n0 + 64 + bn) * DINNER + bseg * 8;

  f32x8 acc[2][2] = {};
  uint4 ra = *(const uint4*)(arow);
  uint4 rb0 = *(const uint4*)(brow0);
  uint4 rb1 = *(const uint4*)(brow1);
  *(uint4*)&sA[0][ar * 32 + aseg * 8] = ra;
  *(uint4*)&sB[0][bn * 32 + bseg * 8] = rb0;
  *(uint4*)&sB[0][(64 + bn) * 32 + bseg * 8] = rb1;
  __syncthreads();

  const int STEPS = DINNER / 32;
  for (int s = 0; s < STEPS; ++s) {
    int cur = s & 1;
    if (s + 1 < STEPS) {
      int k0 = (s + 1) * 32;
      ra  = *(const uint4*)(arow + k0);
      rb0 = *(const uint4*)(brow0 + k0);
      rb1 = *(const uint4*)(brow1 + k0);
    }
#pragma unroll
    for (int i = 0; i < 2; ++i) {
      bf16x16 a = frag_a(sA[cur], wm * 32 + i * 16, 32, 0);
#pragma unroll
      for (int j = 0; j < 2; ++j) {
        bf16x16 bb = frag_b_n(sB[cur], wn * 32 + j * 16, 32, 0);
        acc[i][j] = wmma_bf16(a, bb, acc[i][j]);
      }
    }
    if (s + 1 < STEPS) {
      int nx = 1 - cur;
      *(uint4*)&sA[nx][ar * 32 + aseg * 8] = ra;
      *(uint4*)&sB[nx][bn * 32 + bseg * 8] = rb0;
      *(uint4*)&sB[nx][(64 + bn) * 32 + bseg * 8] = rb1;
    }
    __syncthreads();
  }
  int l = lane_id(); int mlo = (l < 16) ? 0 : 8, nb = l & 15;
#pragma unroll
  for (int i = 0; i < 2; ++i)
#pragma unroll
    for (int j = 0; j < 2; ++j)
#pragma unroll
      for (int r = 0; r < 8; ++r) {
        int gm = m0 + wm * 32 + i * 16 + mlo + r;
        int gn = n0 + wn * 32 + j * 16 + nb;
        out[(size_t)gm * DMODEL + gn] = acc[i][j][r];
      }
}

// CB[l,s] = sum_n C[l,n]*B[s,n] per (b,chunk); M=N=256, K=128
__global__ void k_cb(const u16* __restrict__ bcb, float* __restrict__ cb) {
  __shared__ __align__(16) u16 sA[64 * 32];
  __shared__ __align__(16) u16 sB[128 * 32];
  int b = blockIdx.z >> 3, c = blockIdx.z & 7;
  int m0 = blockIdx.x * 64, n0 = blockIdx.y * 128;
  size_t rb = (size_t)(b * LL + c * CHUNKL) * 256;
  int wv = wave_id(), wm = wv >> 2, wn = wv & 3;
  int tid = threadIdx.x, r4 = tid >> 2, seg = tid & 3;
  f32x8 acc[2][2] = {};
  for (int k0 = 0; k0 < DSTATE; k0 += 32) {
    *(uint4*)&sA[r4 * 32 + seg * 8] =
        *(const uint4*)(bcb + rb + (size_t)(m0 + r4) * 256 + 128 + k0 + seg * 8);        // Cm
    *(uint4*)&sB[r4 * 32 + seg * 8] =
        *(const uint4*)(bcb + rb + (size_t)(n0 + r4) * 256 + k0 + seg * 8);              // Bm
    *(uint4*)&sB[(64 + r4) * 32 + seg * 8] =
        *(const uint4*)(bcb + rb + (size_t)(n0 + 64 + r4) * 256 + k0 + seg * 8);
    __syncthreads();
#pragma unroll
    for (int i = 0; i < 2; ++i) {
      bf16x16 a = frag_a(sA, wm * 32 + i * 16, 32, 0);
#pragma unroll
      for (int j = 0; j < 2; ++j) {
        bf16x16 bb = frag_b_n(sB, wn * 32 + j * 16, 32, 0);
        acc[i][j] = wmma_bf16(a, bb, acc[i][j]);
      }
    }
    __syncthreads();
  }
  int l = lane_id(); int mlo = (l < 16) ? 0 : 8, nb = l & 15;
  float* cbo = cb + (size_t)(b * NCHUNK + c) * CHUNKL * CHUNKL;
#pragma unroll
  for (int i = 0; i < 2; ++i)
#pragma unroll
    for (int j = 0; j < 2; ++j)
#pragma unroll
      for (int r = 0; r < 8; ++r) {
        int gl = m0 + wm * 32 + i * 16 + mlo + r;
        int gs = n0 + wn * 32 + j * 16 + nb;
        cbo[(size_t)gl * CHUNKL + gs] = acc[i][j][r];
      }
}

// Y_diag: per (b,c,h)  M=256(l), N=64(p), K=256(s); A computed with causal decay mask
__global__ void k_ydiag(const float* __restrict__ cb, const float* __restrict__ cA,
                        const u16* __restrict__ xsb, float* __restrict__ Y) {
  __shared__ __align__(16) u16 sA[64 * 32];
  __shared__ __align__(16) u16 sB[32 * 64];
  int z = blockIdx.z; int b = z >> 7, c = (z >> 4) & 7, h = z & 15;
  int m0 = blockIdx.x * 64;
  const float* ca = cA + (size_t)(b * NHEADS + h) * LL + c * CHUNKL;
  const float* cbm = cb + (size_t)(b * NCHUNK + c) * CHUNKL * CHUNKL;
  int wv = wave_id(), wm = wv >> 1, wn = wv & 1;
  int tid = threadIdx.x, r8 = tid >> 3, seg8 = tid & 7;
  f32x8 acc[2] = {};
  for (int k0 = 0; k0 < CHUNKL; k0 += 32) {
    for (int i = tid; i < 64 * 32; i += 256) {
      int r = i >> 5, cc = i & 31;
      int ll = m0 + r, ss = k0 + cc;
      float v = 0.f;
      if (ll >= ss) v = cbm[(size_t)ll * CHUNKL + ss] * expf(ca[ll] - ca[ss]);
      sA[i] = f2bf(v);
    }
    *(uint4*)&sB[r8 * 64 + seg8 * 8] =
        *(const uint4*)(xsb + (size_t)(b * LL + c * CHUNKL + k0 + r8) * DINNER + h * HEADDIM + seg8 * 8);
    if (k0 + 32 < CHUNKL) {
      __builtin_prefetch(xsb + (size_t)(b * LL + c * CHUNKL + k0 + 32 + r8) * DINNER + h * HEADDIM + seg8 * 8, 0, 1);
      __builtin_prefetch(cbm + (size_t)(m0 + (tid >> 5)) * CHUNKL + k0 + 32 + (tid & 31), 0, 1);
    }
    __syncthreads();
    bf16x16 a = frag_a(sA, wm * 16, 32, 0);
#pragma unroll
    for (int j = 0; j < 2; ++j) {
      bf16x16 bb = frag_b_k(sB, 0, 64, wn * 32 + j * 16);
      acc[j] = wmma_bf16(a, bb, acc[j]);
    }
    __syncthreads();
  }
  int l = lane_id(); int mlo = (l < 16) ? 0 : 8, nb = l & 15;
#pragma unroll
  for (int j = 0; j < 2; ++j)
#pragma unroll
    for (int r = 0; r < 8; ++r) {
      int gl = m0 + wm * 16 + mlo + r;
      int gp = wn * 32 + j * 16 + nb;
      Y[(size_t)(b * LL + c * CHUNKL + gl) * DINNER + h * HEADDIM + gp] = acc[j][r];
    }
}

// states: per (b,c,h)  M=64(p), N=128(n), K=256(l); A = decay-scaled X^T
__global__ void k_states(const float* __restrict__ xs, const u16* __restrict__ bcb,
                         const float* __restrict__ cA, float* __restrict__ st) {
  __shared__ __align__(16) u16 sA[64 * 32];
  __shared__ __align__(16) u16 sB[32 * 128];
  int z = blockIdx.z; int b = z >> 7, c = (z >> 4) & 7, h = z & 15;
  const float* ca = cA + (size_t)(b * NHEADS + h) * LL + c * CHUNKL;
  float calast = ca[CHUNKL - 1];
  int wv = wave_id(), wm = wv >> 2, wn = wv & 3;
  int tid = threadIdx.x;
  f32x8 acc[2][2] = {};
  for (int k0 = 0; k0 < CHUNKL; k0 += 32) {
    for (int i = tid; i < 64 * 32; i += 256) {
      int pp = i & 63, lc = i >> 6;
      float d = expf(calast - ca[k0 + lc]);
      sA[pp * 32 + lc] = f2bf(xs[(size_t)(b * LL + c * CHUNKL + k0 + lc) * DINNER + h * HEADDIM + pp] * d);
    }
#pragma unroll
    for (int q = 0; q < 2; ++q) {
      int idx = tid + q * 256, rr = idx >> 4, sg = idx & 15;
      *(uint4*)&sB[rr * 128 + sg * 8] =
          *(const uint4*)(bcb + (size_t)(b * LL + c * CHUNKL + k0 + rr) * 256 + sg * 8);  // Bm
      if (k0 + 32 < CHUNKL)
        __builtin_prefetch(bcb + (size_t)(b * LL + c * CHUNKL + k0 + 32 + rr) * 256 + sg * 8, 0, 1);
    }
    __syncthreads();
#pragma unroll
    for (int i = 0; i < 2; ++i) {
      bf16x16 a = frag_a(sA, wm * 32 + i * 16, 32, 0);
#pragma unroll
      for (int j = 0; j < 2; ++j) {
        bf16x16 bb = frag_b_k(sB, 0, 128, wn * 32 + j * 16);
        acc[i][j] = wmma_bf16(a, bb, acc[i][j]);
      }
    }
    __syncthreads();
  }
  int l = lane_id(); int mlo = (l < 16) ? 0 : 8, nb = l & 15;
  float* sto = st + (size_t)((b * NCHUNK + c) * NHEADS + h) * HEADDIM * DSTATE;
#pragma unroll
  for (int i = 0; i < 2; ++i)
#pragma unroll
    for (int j = 0; j < 2; ++j)
#pragma unroll
      for (int r = 0; r < 8; ++r) {
        int gp = wm * 32 + i * 16 + mlo + r;
        int gn = wn * 32 + j * 16 + nb;
        sto[(size_t)gp * DSTATE + gn] = acc[i][j][r];
      }
}

// Y_off: per (b,c,h)  M=256(l), N=64(p), K=128(n); epilogue *= exp(cA[l]), accumulate into Y
__global__ void k_yoff(const u16* __restrict__ bcb, const u16* __restrict__ nsb,
                       const float* __restrict__ cA, float* __restrict__ Y) {
  __shared__ __align__(16) u16 sA[64 * 32];
  __shared__ __align__(16) u16 sB[64 * 32];
  int z = blockIdx.z; int b = z >> 7, c = (z >> 4) & 7, h = z & 15;
  int m0 = blockIdx.x * 64;
  const float* ca = cA + (size_t)(b * NHEADS + h) * LL + c * CHUNKL;
  const u16* nsp = nsb + (size_t)((b * NCHUNK + c) * NHEADS + h) * HEADDIM * DSTATE;
  int wv = wave_id(), wm = wv >> 1, wn = wv & 1;
  int tid = threadIdx.x, r4 = tid >> 2, seg = tid & 3;
  f32x8 acc[2] = {};
  for (int k0 = 0; k0 < DSTATE; k0 += 32) {
    *(uint4*)&sA[r4 * 32 + seg * 8] =
        *(const uint4*)(bcb + (size_t)(b * LL + c * CHUNKL + m0 + r4) * 256 + 128 + k0 + seg * 8); // Cm
    *(uint4*)&sB[r4 * 32 + seg * 8] =
        *(const uint4*)(nsp + (size_t)r4 * DSTATE + k0 + seg * 8);                                  // ns [p][n]
    __syncthreads();
    bf16x16 a = frag_a(sA, wm * 16, 32, 0);
#pragma unroll
    for (int j = 0; j < 2; ++j) {
      bf16x16 bb = frag_b_n(sB, wn * 32 + j * 16, 32, 0);
      acc[j] = wmma_bf16(a, bb, acc[j]);
    }
    __syncthreads();
  }
  int l = lane_id(); int mlo = (l < 16) ? 0 : 8, nb = l & 15;
#pragma unroll
  for (int j = 0; j < 2; ++j)
#pragma unroll
    for (int r = 0; r < 8; ++r) {
      int gl = m0 + wm * 16 + mlo + r;
      int gp = wn * 32 + j * 16 + nb;
      size_t oi = (size_t)(b * LL + c * CHUNKL + gl) * DINNER + h * HEADDIM + gp;
      Y[oi] += expf(ca[gl]) * acc[j][r];
    }
}

// final: out[b,n,l] = x + proj_b[n] + sum_k A[(b,l),k]*pw[n,k];  A = [fwd+t | bwd_unflip+t]
__global__ void k_final(const float* __restrict__ fwd, const float* __restrict__ bwd,
                        const float* __restrict__ t, const float* __restrict__ xin,
                        const u16* __restrict__ pwb, const float* __restrict__ pb,
                        float* __restrict__ out) {
  __shared__ __align__(16) u16 sA[64 * 32];
  __shared__ __align__(16) u16 sB[128 * 32];
  int m0 = blockIdx.x * 64, n0 = blockIdx.y * 128;
  int wv = wave_id(), wm = wv >> 2, wn = wv & 3;
  int tid = threadIdx.x, r4 = tid >> 2, seg = tid & 3;
  f32x8 acc[2][2] = {};
  for (int k0 = 0; k0 < 2 * DMODEL; k0 += 32) {
    for (int i = tid; i < 64 * 32; i += 256) {
      int r = i >> 5, cc = i & 31;
      int gm = m0 + r, gk = k0 + cc;
      int b = gm >> 11, l = gm & (LL - 1);
      float v;
      if (gk < DMODEL) {
        v = fwd[(size_t)gm * DMODEL + gk] + t[(size_t)gm * DMODEL + gk];
      } else {
        int kk = gk - DMODEL;
        v = bwd[(size_t)(b * LL + (LL - 1 - l)) * DMODEL + kk] + t[(size_t)gm * DMODEL + kk];
      }
      sA[i] = f2bf(v);
    }
    *(uint4*)&sB[r4 * 32 + seg * 8] =
        *(const uint4*)(pwb + (size_t)(n0 + r4) * (2 * DMODEL) + k0 + seg * 8);
    *(uint4*)&sB[(64 + r4) * 32 + seg * 8] =
        *(const uint4*)(pwb + (size_t)(n0 + 64 + r4) * (2 * DMODEL) + k0 + seg * 8);
    __syncthreads();
#pragma unroll
    for (int i = 0; i < 2; ++i) {
      bf16x16 a = frag_a(sA, wm * 32 + i * 16, 32, 0);
#pragma unroll
      for (int j = 0; j < 2; ++j) {
        bf16x16 bb = frag_b_n(sB, wn * 32 + j * 16, 32, 0);
        acc[i][j] = wmma_bf16(a, bb, acc[i][j]);
      }
    }
    __syncthreads();
  }
  int l = lane_id(); int mlo = (l < 16) ? 0 : 8, nb = l & 15;
#pragma unroll
  for (int i = 0; i < 2; ++i)
#pragma unroll
    for (int j = 0; j < 2; ++j)
#pragma unroll
      for (int r = 0; r < 8; ++r) {
        int gm = m0 + wm * 32 + i * 16 + mlo + r;
        int gn = n0 + wn * 32 + j * 16 + nb;
        int b = gm >> 11, ll = gm & (LL - 1);
        size_t oi = ((size_t)b * DMODEL + gn) * LL + ll;
        out[oi] = xin[oi] + pb[gn] + acc[i][j][r];
      }
}

// ---------------- host launcher ----------------

extern "C" void kernel_launch(void* const* d_in, const int* in_sizes, int n_in,
                              void* d_out, int out_size, void* d_ws, size_t ws_size,
                              hipStream_t stream) {
  (void)in_sizes; (void)n_in; (void)out_size; (void)ws_size;
  const float* x   = (const float*)d_in[0];
  const float* gnw = (const float*)d_in[1];
  const float* gnb = (const float*)d_in[2];
  const float* pw  = (const float*)d_in[3];
  const float* pb  = (const float*)d_in[4];

  float* W   = (float*)d_ws;
  float* t   = W;
  float* zx  = t   + (size_t)MTOT * DMODEL;
  float* xbc = zx  + (size_t)MTOT * DINPROJ;
  float* dtb = xbc + (size_t)MTOT * CONVDIM;
  float* ca  = dtb + (size_t)MTOT * NHEADS;
  float* xs  = ca  + (size_t)BB * NHEADS * LL;
  float* cb  = xs  + (size_t)MTOT * DINNER;
  float* st  = cb  + (size_t)BB * NCHUNK * CHUNKL * CHUNKL;
  float* Y   = st  + (size_t)BB * NCHUNK * NHEADS * HEADDIM * DSTATE;
  float* fwd = Y   + (size_t)MTOT * DINNER;
  float* bwd = fwd + (size_t)MTOT * DMODEL;
  float* stats = bwd + (size_t)MTOT * DMODEL;

  u16* BF   = (u16*)(stats + 8);
  u16* tb   = BF;
  u16* inwb = tb   + (size_t)MTOT * DMODEL;
  u16* owb  = inwb + (size_t)DINPROJ * DMODEL;
  u16* pwb  = owb  + (size_t)DMODEL * DINNER;
  u16* bcb  = pwb  + (size_t)DMODEL * 2 * DMODEL;
  u16* xsb  = bcb  + (size_t)MTOT * 256;
  u16* ynb  = xsb  + (size_t)MTOT * DINNER;
  u16* nsb  = ynb  + (size_t)MTOT * DINNER;

  k_gn_stats<<<BB, 256, 0, stream>>>(x, stats);
  k_normT<<<(MTOT * DMODEL) / 256, 256, 0, stream>>>(x, stats, gnw, gnb, t, tb);
  k_tobf16<<<(DMODEL * 2 * DMODEL + 255) / 256, 256, 0, stream>>>(pw, pwb, DMODEL * 2 * DMODEL);

  for (int dir = 0; dir < 2; ++dir) {
    const float* inw    = (const float*)d_in[5 + dir * 8 + 0];
    const float* cw     = (const float*)d_in[5 + dir * 8 + 1];
    const float* cbias  = (const float*)d_in[5 + dir * 8 + 2];
    const float* dtbias = (const float*)d_in[5 + dir * 8 + 3];
    const float* alog   = (const float*)d_in[5 + dir * 8 + 4];
    const float* Dh     = (const float*)d_in[5 + dir * 8 + 5];
    const float* nw     = (const float*)d_in[5 + dir * 8 + 6];
    const float* ow     = (const float*)d_in[5 + dir * 8 + 7];
    float* odir = dir ? bwd : fwd;

    k_tobf16<<<(DINPROJ * DMODEL + 255) / 256, 256, 0, stream>>>(inw, inwb, DINPROJ * DMODEL);
    k_tobf16<<<(DMODEL * DINNER + 255) / 256, 256, 0, stream>>>(ow, owb, DMODEL * DINNER);

    k_in_proj<<<dim3(MTOT / 64, (DINPROJ + 127) / 128), 256, 0, stream>>>(tb, inwb, zx, dir);
    k_conv<<<(MTOT * CONVDIM) / 256, 256, 0, stream>>>(zx, cw, cbias, xbc, bcb);
    k_dt_scan<<<BB * NHEADS * NCHUNK, 256, 0, stream>>>(zx, dtbias, alog, dtb, ca);
    k_xs<<<(MTOT * DINNER) / 256, 256, 0, stream>>>(xbc, dtb, xs, xsb);
    k_cb<<<dim3(CHUNKL / 64, CHUNKL / 128, BB * NCHUNK), 256, 0, stream>>>(bcb, cb);
    k_ydiag<<<dim3(CHUNKL / 64, 1, BB * NCHUNK * NHEADS), 256, 0, stream>>>(cb, ca, xsb, Y);
    k_states<<<dim3(1, 1, BB * NCHUNK * NHEADS), 256, 0, stream>>>(xs, bcb, ca, st);
    k_chunkrec<<<(BB * NHEADS * HEADDIM * DSTATE) / 256, 256, 0, stream>>>(st, ca, nsb);
    k_yoff<<<dim3(CHUNKL / 64, 1, BB * NCHUNK * NHEADS), 256, 0, stream>>>(bcb, nsb, ca, Y);
    k_gate_rms<<<MTOT, 256, 0, stream>>>(Y, xbc, zx, Dh, nw, ynb);
    k_outproj<<<dim3(MTOT / 64, DMODEL / 128), 256, 0, stream>>>(ynb, owb, odir);
  }

  k_final<<<dim3(MTOT / 64, DMODEL / 128), 256, 0, stream>>>(fwd, bwd, t, x, pwb, pb, (float*)d_out);
}
